// SBlock_43207370997899
// MI455X (gfx1250) — compile-verified, matrix operand order checked by reading, not run on hardware
//
#include <hip/hip_runtime.h>
#include <hip/hip_bf16.h>

typedef __attribute__((ext_vector_type(16))) __bf16 v16bf;
typedef __attribute__((ext_vector_type(8)))  float  v8f;

#define NN   50000
#define EE   800000
#define DIM  128
#define HH   4
#define CCH  32
#define EDIM 16
#define FFD  512

// ---------------------------------------------------------------- utilities

__global__ __launch_bounds__(256) void k_fill_u32(unsigned int* p, unsigned int v, int n) {
    int i = blockIdx.x * blockDim.x + threadIdx.x;
    if (i < n) p[i] = v;
}

__global__ __launch_bounds__(256) void k_f32_to_bf16(const float* __restrict__ s,
                                                     __bf16* __restrict__ d, int n) {
    int i = blockIdx.x * blockDim.x + threadIdx.x;
    if (i < n) d[i] = (__bf16)s[i];
}

// Wt[n*K + k] = W[k*N + n]   (weights transposed so the WMMA B-fragment
// loads contiguous bf16 spans along K, same as the A-fragment)
__global__ __launch_bounds__(256) void k_transpose_bf16(const float* __restrict__ W,
                                                        __bf16* __restrict__ Wt,
                                                        int K, int N) {
    int i = blockIdx.x * blockDim.x + threadIdx.x;
    if (i >= K * N) return;
    int k = i / N, n = i % N;
    Wt[(size_t)n * K + k] = (__bf16)W[i];
}

__device__ inline void atomicMaxF32(float* addr, float val) {
    if (val >= 0.0f) atomicMax((int*)addr, __float_as_int(val));
    else             atomicMin((unsigned int*)addr, (unsigned int)__float_as_int(val));
}

// ---------------------------------------------------------------- WMMA GEMM
// C[M,N] = A[M,K] @ B[K,N] + bias; A bf16 row-major, B supplied as Bt[N,K].
// One wave32 computes a 16 x (16*NSUB) strip: the A fragment is loaded once
// per K-step (2 x global_load_b128) and reused by NSUB wmma ops.
//
// A fragment (ISA 7.12.2, 16-bit A 16x32): lane L -> row m = L%16, half = L/16,
//   element i (0..15): K = (i/8)*16 + half*8 + (i%8)  ==> two contiguous
//   8-element (16B) spans at k0+half*8 and k0+16+half*8.
// K must be a multiple of 32 (all launches use K = 128 or 512).

union ABFrag { v16bf v; uint4 q2[2]; };

template <int NSUB>
__global__ __launch_bounds__(256)
void gemm_bf16_wmma(const __bf16* __restrict__ A, const __bf16* __restrict__ Bt,
                    const float* __restrict__ bias,
                    float* __restrict__ Cf, __bf16* __restrict__ Cbf,
                    int M, int K, int N, int relu)
{
    const int lane = threadIdx.x & 31;
    const int wave = (blockIdx.x * blockDim.x + threadIdx.x) >> 5;
    const int nStrips = N / (16 * NSUB);
    const int total   = (M >> 4) * nStrips;
    if (wave >= total) return;
    const int m0   = (wave / nStrips) << 4;
    const int n0   = (wave % nStrips) * (16 * NSUB);
    const int half = lane >> 4;
    const int l15  = lane & 15;

    const __bf16* Arow = A  + (size_t)(m0 + l15) * K;
    const __bf16* Brow = Bt + (size_t)(n0 + l15) * K;   // sub-tile s adds s*16*K

    v8f acc[NSUB];
#pragma unroll
    for (int s = 0; s < NSUB; ++s) acc[s] = (v8f){};

    for (int k0 = 0; k0 < K; k0 += 32) {
        ABFrag a;
        a.q2[0] = *(const uint4*)(Arow + k0 + half * 8);
        a.q2[1] = *(const uint4*)(Arow + k0 + 16 + half * 8);
#pragma unroll
        for (int s = 0; s < NSUB; ++s) {
            ABFrag b;
            const __bf16* Bs = Brow + (size_t)s * 16 * K;
            b.q2[0] = *(const uint4*)(Bs + k0 + half * 8);
            b.q2[1] = *(const uint4*)(Bs + k0 + 16 + half * 8);
            acc[s] = __builtin_amdgcn_wmma_f32_16x16x32_bf16(false, a.v, false, b.v,
                                                             (short)0, acc[s], false, false);
        }
    }

#pragma unroll
    for (int s = 0; s < NSUB; ++s) {
        const int col = n0 + s * 16 + l15;
        const float bv = bias ? bias[col] : 0.0f;
#pragma unroll
        for (int r = 0; r < 8; ++r) {        // C/D layout: VGPR r -> row half*8 + r
            const int row = m0 + (half << 3) + r;
            float val = acc[s][r] + bv;
            if (relu && val < 0.0f) val = 0.0f;
            if (Cf)  Cf [(size_t)row * N + col] = val;
            if (Cbf) Cbf[(size_t)row * N + col] = (__bf16)val;
        }
    }
}

// ---------------------------------------------------------------- edge phase
// wave32 per edge; lane owns channels c0..c0+3 (c0 = lane*4); head = lane/8.

__device__ inline float4 edge_feat(const float* __restrict__ eaRow,
                                   const float* __restrict__ We,
                                   const float* __restrict__ be, int c0) {
    float4 ev = *(const float4*)(be + c0);
#pragma unroll
    for (int j = 0; j < EDIM; ++j) {
        float w = eaRow[j];
        const float4 wv = *(const float4*)(We + j * DIM + c0);
        ev.x += w * wv.x; ev.y += w * wv.y; ev.z += w * wv.z; ev.w += w * wv.w;
    }
    return ev;
}

__global__ __launch_bounds__(256)
void edge_alpha_max(const long long* __restrict__ ei,
                    const float* __restrict__ q, const float* __restrict__ k,
                    const float* __restrict__ ea,
                    const float* __restrict__ We, const float* __restrict__ be,
                    float* __restrict__ alpha, float* __restrict__ m)
{
    const int e    = (blockIdx.x * blockDim.x + threadIdx.x) >> 5;
    const int lane = threadIdx.x & 31;
    if (e >= EE) return;
    const long long s = ei[e], d = ei[EE + e];
    const int c0 = lane * 4;

    const float4 qv = *(const float4*)(q + d * DIM + c0);
    const float4 kv = *(const float4*)(k + s * DIM + c0);
    const float4 ev = edge_feat(ea + (size_t)e * EDIM, We, be, c0);

    float sum = qv.x * (kv.x + ev.x) + qv.y * (kv.y + ev.y)
              + qv.z * (kv.z + ev.z) + qv.w * (kv.w + ev.w);
    sum += __shfl_xor(sum, 1, 32);
    sum += __shfl_xor(sum, 2, 32);
    sum += __shfl_xor(sum, 4, 32);
    sum *= 0.17677669529663687f;             // 1/sqrt(32)

    const int h = lane >> 3;
    if ((lane & 7) == 0) {
        alpha[(size_t)e * HH + h] = sum;
        atomicMaxF32(m + d * HH + h, sum);
    }
}

__global__ __launch_bounds__(256)
void edge_accum(const long long* __restrict__ ei,
                const float* __restrict__ v, const float* __restrict__ ea,
                const float* __restrict__ We, const float* __restrict__ be,
                const float* __restrict__ alpha, const float* __restrict__ m,
                float* __restrict__ denom, float* __restrict__ outun)
{
    const int e    = (blockIdx.x * blockDim.x + threadIdx.x) >> 5;
    const int lane = threadIdx.x & 31;
    if (e >= EE) return;
    const long long s = ei[e], d = ei[EE + e];
    const int h  = lane >> 3;
    const int c0 = lane * 4;

    const float a = __expf(alpha[(size_t)e * HH + h] - m[d * HH + h]);
    if ((lane & 7) == 0) atomicAdd(denom + d * HH + h, a);

    const float4 vv = *(const float4*)(v + s * DIM + c0);
    const float4 ev = edge_feat(ea + (size_t)e * EDIM, We, be, c0);
    float* o = outun + d * DIM + c0;
    atomicAdd(o + 0, a * (vv.x + ev.x));
    atomicAdd(o + 1, a * (vv.y + ev.y));
    atomicAdd(o + 2, a * (vv.z + ev.z));
    atomicAdd(o + 3, a * (vv.w + ev.w));
}

// ---------------------------------------------------------------- node phase

__device__ inline float waveSum32(float x) {
    x += __shfl_xor(x, 1, 32);  x += __shfl_xor(x, 2, 32);
    x += __shfl_xor(x, 4, 32);  x += __shfl_xor(x, 8, 32);
    x += __shfl_xor(x, 16, 32);
    return x;
}

__global__ __launch_bounds__(256)
void node_beta_ln1(const float* __restrict__ x,
                   const float* __restrict__ outun, const float* __restrict__ denom,
                   const float* __restrict__ xr, const float* __restrict__ Wb,
                   const float* __restrict__ g1, const float* __restrict__ b1,
                   float* __restrict__ hf, __bf16* __restrict__ hbf)
{
    const int n    = (blockIdx.x * blockDim.x + threadIdx.x) >> 5;
    const int lane = threadIdx.x & 31;
    if (n >= NN) return;
    const int c0 = lane * 4;
    const int h  = lane >> 3;

    const float den = denom[(size_t)n * HH + h] + 1e-16f;
    float4 o = *(const float4*)(outun + (size_t)n * DIM + c0);
    o.x /= den; o.y /= den; o.z /= den; o.w /= den;
    const float4 r = *(const float4*)(xr + (size_t)n * DIM + c0);

    // beta = sigmoid([out, xr, out-xr] @ Wbeta)
    float loc = o.x * Wb[c0+0] + o.y * Wb[c0+1] + o.z * Wb[c0+2] + o.w * Wb[c0+3]
              + r.x * Wb[DIM+c0+0] + r.y * Wb[DIM+c0+1] + r.z * Wb[DIM+c0+2] + r.w * Wb[DIM+c0+3]
              + (o.x-r.x) * Wb[2*DIM+c0+0] + (o.y-r.y) * Wb[2*DIM+c0+1]
              + (o.z-r.z) * Wb[2*DIM+c0+2] + (o.w-r.w) * Wb[2*DIM+c0+3];
    const float beta = 1.0f / (1.0f + __expf(-waveSum32(loc)));

    const float4 xv = *(const float4*)(x + (size_t)n * DIM + c0);
    float y[4];
    y[0] = xv.x + beta * r.x + (1.0f - beta) * o.x;
    y[1] = xv.y + beta * r.y + (1.0f - beta) * o.y;
    y[2] = xv.z + beta * r.z + (1.0f - beta) * o.z;
    y[3] = xv.w + beta * r.w + (1.0f - beta) * o.w;

    const float sm = waveSum32(y[0] + y[1] + y[2] + y[3]);
    const float sq = waveSum32(y[0]*y[0] + y[1]*y[1] + y[2]*y[2] + y[3]*y[3]);
    const float mu  = sm * (1.0f / DIM);
    const float inv = rsqrtf(sq * (1.0f / DIM) - mu * mu + 1e-5f);
#pragma unroll
    for (int t = 0; t < 4; ++t) {
        const float hv = (y[t] - mu) * inv * g1[c0 + t] + b1[c0 + t];
        hf [(size_t)n * DIM + c0 + t] = hv;
        hbf[(size_t)n * DIM + c0 + t] = (__bf16)hv;
    }
}

__global__ __launch_bounds__(256)
void node_ln2(const float* __restrict__ hf, const float* __restrict__ ff,
              const float* __restrict__ g2, const float* __restrict__ b2,
              float* __restrict__ out)
{
    const int n    = (blockIdx.x * blockDim.x + threadIdx.x) >> 5;
    const int lane = threadIdx.x & 31;
    if (n >= NN) return;
    const int c0 = lane * 4;
    const float4 hv = *(const float4*)(hf + (size_t)n * DIM + c0);
    const float4 fv = *(const float4*)(ff + (size_t)n * DIM + c0);
    float y[4] = { hv.x + fv.x, hv.y + fv.y, hv.z + fv.z, hv.w + fv.w };
    const float sm = waveSum32(y[0] + y[1] + y[2] + y[3]);
    const float sq = waveSum32(y[0]*y[0] + y[1]*y[1] + y[2]*y[2] + y[3]*y[3]);
    const float mu  = sm * (1.0f / DIM);
    const float inv = rsqrtf(sq * (1.0f / DIM) - mu * mu + 1e-5f);
#pragma unroll
    for (int t = 0; t < 4; ++t)
        out[(size_t)n * DIM + c0 + t] = (y[t] - mu) * inv * g2[c0 + t] + b2[c0 + t];
}

// ---------------------------------------------------------------- launcher

extern "C" void kernel_launch(void* const* d_in, const int* in_sizes, int n_in,
                              void* d_out, int out_size, void* d_ws, size_t ws_size,
                              hipStream_t stream) {
    (void)in_sizes; (void)n_in; (void)out_size; (void)ws_size;

    const float*     x    = (const float*)d_in[0];
    const long long* ei   = (const long long*)d_in[1];   // int64 [2,E]
    const float*     ea   = (const float*)d_in[2];
    const float *Wq = (const float*)d_in[3],  *bq = (const float*)d_in[4];
    const float *Wk = (const float*)d_in[5],  *bk = (const float*)d_in[6];
    const float *Wv = (const float*)d_in[7],  *bv = (const float*)d_in[8];
    const float *We = (const float*)d_in[9],  *be = (const float*)d_in[10];
    const float *Ws = (const float*)d_in[11], *bs = (const float*)d_in[12];
    const float *Wb = (const float*)d_in[13];
    const float *g1 = (const float*)d_in[14], *b1 = (const float*)d_in[15];
    const float *Wf1 = (const float*)d_in[16], *bf1 = (const float*)d_in[17];
    const float *Wf2 = (const float*)d_in[18], *bf2 = (const float*)d_in[19];
    const float *g2 = (const float*)d_in[20], *b2 = (const float*)d_in[21];
    float* out = (float*)d_out;

    // workspace carve-out (256B aligned)
    char* w = (char*)d_ws;
    auto carve = [&](size_t bytes) { void* p = (void*)w; w += (bytes + 255) & ~(size_t)255; return p; };
    __bf16* xbf  = (__bf16*)carve((size_t)NN * DIM * 2);
    __bf16* wqT  = (__bf16*)carve((size_t)DIM * DIM * 2);
    __bf16* wkT  = (__bf16*)carve((size_t)DIM * DIM * 2);
    __bf16* wvT  = (__bf16*)carve((size_t)DIM * DIM * 2);
    __bf16* wsT  = (__bf16*)carve((size_t)DIM * DIM * 2);
    __bf16* wf1T = (__bf16*)carve((size_t)FFD * DIM * 2);
    __bf16* wf2T = (__bf16*)carve((size_t)DIM * FFD * 2);
    float*  qf   = (float*)carve((size_t)NN * DIM * 4);
    float*  kf   = (float*)carve((size_t)NN * DIM * 4);
    float*  vf   = (float*)carve((size_t)NN * DIM * 4);
    float*  xrf  = (float*)carve((size_t)NN * DIM * 4);
    float*  alph = (float*)carve((size_t)EE * HH * 4);
    float*  mbuf = (float*)carve((size_t)NN * HH * 4);
    float*  dbuf = (float*)carve((size_t)NN * HH * 4);
    float*  oun  = (float*)carve((size_t)NN * DIM * 4);
    float*  hf   = (float*)carve((size_t)NN * DIM * 4);
    __bf16* hbf  = (__bf16*)carve((size_t)NN * DIM * 2);
    __bf16* hhbf = (__bf16*)carve((size_t)NN * FFD * 2);
    float*  ffb  = (float*)carve((size_t)NN * DIM * 4);

    // 1. bf16 conversions / weight transposes
    {
        int n = NN * DIM;
        k_f32_to_bf16<<<(n + 255) / 256, 256, 0, stream>>>(x, xbf, n);
        k_transpose_bf16<<<(DIM*DIM + 255) / 256, 256, 0, stream>>>(Wq, wqT, DIM, DIM);
        k_transpose_bf16<<<(DIM*DIM + 255) / 256, 256, 0, stream>>>(Wk, wkT, DIM, DIM);
        k_transpose_bf16<<<(DIM*DIM + 255) / 256, 256, 0, stream>>>(Wv, wvT, DIM, DIM);
        k_transpose_bf16<<<(DIM*DIM + 255) / 256, 256, 0, stream>>>(Ws, wsT, DIM, DIM);
        k_transpose_bf16<<<(DIM*FFD + 255) / 256, 256, 0, stream>>>(Wf1, wf1T, DIM, FFD);
        k_transpose_bf16<<<(FFD*DIM + 255) / 256, 256, 0, stream>>>(Wf2, wf2T, FFD, DIM);
    }

    // 2. projections: q,k,v,xr  ([50000x128] @ [128x128], 16x64 strip per wave)
    {
        const int strips = (NN / 16) * (DIM / 64);      // 6250
        const int blocks = (strips + 7) / 8;
        gemm_bf16_wmma<4><<<blocks, 256, 0, stream>>>(xbf, wqT, bq, qf,  nullptr, NN, DIM, DIM, 0);
        gemm_bf16_wmma<4><<<blocks, 256, 0, stream>>>(xbf, wkT, bk, kf,  nullptr, NN, DIM, DIM, 0);
        gemm_bf16_wmma<4><<<blocks, 256, 0, stream>>>(xbf, wvT, bv, vf,  nullptr, NN, DIM, DIM, 0);
        gemm_bf16_wmma<4><<<blocks, 256, 0, stream>>>(xbf, wsT, bs, xrf, nullptr, NN, DIM, DIM, 0);
    }

    // 3. init segment-softmax state
    k_fill_u32<<<(NN*HH + 255) / 256, 256, 0, stream>>>((unsigned int*)mbuf, 0xFF800000u, NN*HH); // -inf
    k_fill_u32<<<(NN*HH + 255) / 256, 256, 0, stream>>>((unsigned int*)dbuf, 0u, NN*HH);
    k_fill_u32<<<(NN*DIM + 255) / 256, 256, 0, stream>>>((unsigned int*)oun, 0u, NN*DIM);

    // 4. edge passes (1 wave per edge)
    edge_alpha_max<<<EE / 8, 256, 0, stream>>>(ei, qf, kf, ea, We, be, alph, mbuf);
    edge_accum   <<<EE / 8, 256, 0, stream>>>(ei, vf, ea, We, be, alph, mbuf, dbuf, oun);

    // 5. node pass 1: normalize + beta gate + LN1
    node_beta_ln1<<<(NN + 7) / 8, 256, 0, stream>>>(x, oun, dbuf, xrf, Wb, g1, b1, hf, hbf);

    // 6. FFN via WMMA
    {
        const int s1 = (NN / 16) * (FFD / 64);          // 25000
        gemm_bf16_wmma<4><<<(s1 + 7) / 8, 256, 0, stream>>>(hbf, wf1T, bf1, nullptr, hhbf, NN, DIM, FFD, 1);
        const int s2 = (NN / 16) * (DIM / 64);          // 6250
        gemm_bf16_wmma<4><<<(s2 + 7) / 8, 256, 0, stream>>>(hhbf, wf2T, bf2, ffb, nullptr, NN, FFD, DIM, 0);
    }

    // 7. final LN -> output
    node_ln2<<<(NN + 7) / 8, 256, 0, stream>>>(hf, ffb, g2, b2, out);
}
